// BasicMoELayer_7980049236629
// MI455X (gfx1250) — matrix operand order repaired and checked by reading, not compiled, not run on hardware
//
#include <hip/hip_runtime.h>
#include <stdint.h>

#define T_TOK   8192
#define D_MODEL 1024
#define D_FF    4096
#define N_EXP   8

#define TM 32                   // tokens per block tile
#define FN 128                  // d_ff chunk processed per iteration
#define XS_STRIDE (D_MODEL + 8) // pad to avoid LDS bank conflicts
#define HS_STRIDE (FN + 8)

#define XS_BYTES (TM * XS_STRIDE * 2)
#define HS_BYTES (TM * HS_STRIDE * 2)
#define SMEM_BYTES (XS_BYTES + HS_BYTES + TM * 4 + 16)

typedef __attribute__((ext_vector_type(16))) __bf16 v16bf;
typedef __attribute__((ext_vector_type(8)))  float  v8f;

// Fast sigmoid: single v_rcp_f32 instead of the precise-division sequence.
// Result is immediately rounded to bf16, so 1-ulp rcp error is immaterial.
__device__ __forceinline__ float fast_sigmoid(float v) {
  return __builtin_amdgcn_rcpf(1.f + __expf(-v));
}

// ---- WMMA fragment loaders -------------------------------------------------
// A-matrix 16x32 bf16 layout (ISA 7.12.2): lanes 0-15 -> M=0..15 (kh=0),
// lanes 16-31 -> same M, K offset +8 within group; VGPR v: group g=v>>2 adds
// +16, pair (v&3)*2. All K offsets even -> 32-bit LDS loads.
__device__ __forceinline__ v16bf load_A_lds(const __bf16* base, int stride,
                                            int lane, int k0) {
  const int m  = lane & 15;
  const int kh = (lane >> 4) & 1;
  union { v16bf v; uint32_t u[8]; } r;
#pragma unroll
  for (int vv = 0; vv < 8; ++vv) {
    int k = k0 + ((vv >> 2) << 4) + (kh << 3) + ((vv & 3) << 1);
    r.u[vv] = *reinterpret_cast<const uint32_t*>(base + m * stride + k);
  }
  return r.v;
}

// B-matrix 32x16 bf16: lanes 0-15 -> N=0..15; lanes 16-31 -> K offset +8;
// VGPR v holds K pair 16*(v>>2)+8*kh+2*(v&3),+1 (matches A). W is row-major [K][rs] fp32 -> convert inline.
// Each dword load is two contiguous 64B segments across the wave.
__device__ __forceinline__ v16bf load_B_g(const float* __restrict__ W, int rs,
                                          int lane) {
  const int n  = lane & 15;
  const int kh = (lane >> 4) & 1;
  union { v16bf v; __bf16 h[16]; } r;
#pragma unroll
  for (int vv = 0; vv < 8; ++vv) {
    int k = ((vv >> 2) << 4) + (kh << 3) + ((vv & 3) << 1);
    const float* p = W + (size_t)k * rs + n;
    r.h[2 * vv]     = (__bf16)p[0];
    r.h[2 * vv + 1] = (__bf16)p[rs];
  }
  return r.v;
}

// ---- Kernel 0: zero expert counters ---------------------------------------
__global__ void moe_zero_counts(int* __restrict__ counts) {
  if (threadIdx.x < N_EXP) counts[threadIdx.x] = 0;
}

__global__ void moe_zero_out(float* __restrict__ out, int n) {
  int i = blockIdx.x * 256 + threadIdx.x;
  if (i < n) out[i] = 0.f;
}

// ---- Kernel 1: router GEMV + top-2 softmax + scatter ----------------------
// One wave32 per token; router staged in LDS; x reads coalesced across lanes.
__global__ __launch_bounds__(256) void moe_router_scatter(
    const float* __restrict__ x, const float* __restrict__ router,
    int* __restrict__ tlist, int* __restrict__ counts,
    float* __restrict__ gates) {
  __shared__ float Rs[D_MODEL * N_EXP]; // 32KB
  const int tid = threadIdx.x;
  for (int i = tid; i < D_MODEL * N_EXP; i += 256) Rs[i] = router[i];
  __syncthreads();

  const int lane  = tid & 31;
  const int wave  = tid >> 5;
  const int token = blockIdx.x * 8 + wave;

  float acc[N_EXP];
#pragma unroll
  for (int e = 0; e < N_EXP; ++e) acc[e] = 0.f;

  const float* xr = x + (size_t)token * D_MODEL;
  for (int j = 0; j < D_MODEL / 32; ++j) {
    float xv = xr[j * 32 + lane];
    const float* rrow = &Rs[(j * 32 + lane) * N_EXP];
#pragma unroll
    for (int e = 0; e < N_EXP; ++e) acc[e] += xv * rrow[e];
  }
#pragma unroll
  for (int off = 16; off > 0; off >>= 1)
#pragma unroll
    for (int e = 0; e < N_EXP; ++e) acc[e] += __shfl_xor(acc[e], off, 32);

  if (lane == 0) {
    int i0 = 0; float v0 = acc[0];
#pragma unroll
    for (int e = 1; e < N_EXP; ++e)
      if (acc[e] > v0) { v0 = acc[e]; i0 = e; }
    int i1 = (i0 == 0) ? 1 : 0; float v1 = acc[i1];
#pragma unroll
    for (int e = 0; e < N_EXP; ++e)
      if (e != i0 && acc[e] > v1) { v1 = acc[e]; i1 = e; }
    // softmax over {v0, v1}, v0 >= v1
    float t = __expf(v1 - v0);
    float inv = __builtin_amdgcn_rcpf(1.f + t);
    gates[token * 2 + 0] = inv;
    gates[token * 2 + 1] = t * inv;
    int p0 = atomicAdd(&counts[i0], 1);
    tlist[i0 * T_TOK + p0] = (token << 1);
    int p1 = atomicAdd(&counts[i1], 1);
    tlist[i1 * T_TOK + p1] = (token << 1) | 1;
  }
}

// ---- Kernel 2: fused expert FFN (x@W1 -> swish -> @W2), WMMA bf16 ---------
// Block: 256 threads = 8 waves; 32 gathered tokens per block (halves L2
// weight re-reads vs TM=16). Wave w: row half mh = w>>2, column quad q = w&3.
//   GEMM1: computes h rows [16mh,16mh+16) x cols [32q, 32q+32) of each chunk.
//   GEMM2: accumulates out rows [16mh,+16) x cols [256q, +256) in registers
//          (16 x v8f = 128 VGPRs/lane).
__global__ __launch_bounds__(256) void moe_expert_ffn(
    const float* __restrict__ x, const float* __restrict__ W1,
    const float* __restrict__ W2, const int* __restrict__ tlist,
    const int* __restrict__ counts, const float* __restrict__ gates,
    float* __restrict__ dst0, float* __restrict__ dst1, int atomic_mode) {
  extern __shared__ __align__(16) char smem[];
  __bf16* Xs   = (__bf16*)smem;                         // TM x XS_STRIDE
  __bf16* Hs   = (__bf16*)(smem + XS_BYTES);            // TM x HS_STRIDE
  int*    toks = (int*)(smem + XS_BYTES + HS_BYTES);    // TM

  const int e    = blockIdx.y;
  const int tile = blockIdx.x;
  const int cnt  = counts[e];
  const int row0 = tile * TM;
  if (row0 >= cnt) return; // uniform per block (EXEC stays all-1 for WMMA)

  const int tid  = threadIdx.x;
  const int lane = tid & 31;
  const int wave = tid >> 5;

  if (tid < TM) {
    int r = row0 + tid;
    toks[tid] = tlist[e * T_TOK + (r < cnt ? r : row0)]; // pad rows reuse row0
  }
  __syncthreads();

  // Stage gathered X tile as bf16 (float4 loads, coalesced along columns)
  for (int j = tid; j < TM * D_MODEL / 4; j += 256) {
    int m = j >> 8;            // / (D_MODEL/4)
    int c = (j & 255) << 2;
    int tok = toks[m] >> 1;
    float4 v = *(const float4*)(x + (size_t)tok * D_MODEL + c);
    __bf16* dst = Xs + m * XS_STRIDE + c;
    dst[0] = (__bf16)v.x; dst[1] = (__bf16)v.y;
    dst[2] = (__bf16)v.z; dst[3] = (__bf16)v.w;
  }
  __syncthreads();

  const float* W1e = W1 + (size_t)e * D_MODEL * D_FF;
  const float* W2e = W2 + (size_t)e * D_FF * D_MODEL;

  v8f oacc[16];
#pragma unroll
  for (int t = 0; t < 16; ++t)
    oacc[t] = (v8f){0.f, 0.f, 0.f, 0.f, 0.f, 0.f, 0.f, 0.f};

  const int mrow0 = (wave >> 2) << 4;  // 0 or 16: row half
  const int quad  = wave & 3;
  const int hn0   = quad * 32;         // h column strip within chunk (width 32)
  const int on0   = quad * 256;        // out column strip (width 256)
  const __bf16* Xa = Xs + mrow0 * XS_STRIDE;
  const __bf16* Ha = Hs + mrow0 * HS_STRIDE;

  for (int ffc = 0; ffc < D_FF; ffc += FN) {
    // ---- GEMM1: h(16 x 32) over K = D_MODEL --------------------------------
    v8f hacc0 = (v8f){0.f, 0.f, 0.f, 0.f, 0.f, 0.f, 0.f, 0.f};
    v8f hacc1 = (v8f){0.f, 0.f, 0.f, 0.f, 0.f, 0.f, 0.f, 0.f};
#pragma unroll 2
    for (int kk = 0; kk < D_MODEL; kk += 32) {
      __builtin_prefetch(W1e + (size_t)(kk + 32) * D_FF + ffc + hn0, 0, 1);
      v16bf a  = load_A_lds(Xa, XS_STRIDE, lane, kk);
      v16bf b0 = load_B_g(W1e + (size_t)kk * D_FF + (ffc + hn0), D_FF, lane);
      hacc0 = __builtin_amdgcn_wmma_f32_16x16x32_bf16(
          false, a, false, b0, (short)0, hacc0, false, false);
      v16bf b1 = load_B_g(W1e + (size_t)kk * D_FF + (ffc + hn0 + 16), D_FF,
                          lane);
      hacc1 = __builtin_amdgcn_wmma_f32_16x16x32_bf16(
          false, a, false, b1, (short)0, hacc1, false, false);
    }
    __syncthreads(); // previous chunk's Hs consumers are done (WAR)
    {
      const int n  = lane & 15;
      const int mb = mrow0 + ((lane >> 4) << 3);
#pragma unroll
      for (int r = 0; r < 8; ++r) { // C layout: VGPR r -> M = r + 8*(lane>=16)
        float v0 = hacc0[r];
        Hs[(mb + r) * HS_STRIDE + hn0 + n] = (__bf16)(v0 * fast_sigmoid(v0));
        float v1 = hacc1[r];
        Hs[(mb + r) * HS_STRIDE + hn0 + 16 + n] =
            (__bf16)(v1 * fast_sigmoid(v1));
      }
    }
    __syncthreads();
    // ---- GEMM2: out(16 x 256) += H(16 x FN) @ W2[ffc:ffc+FN, on0:on0+256] --
#pragma unroll
    for (int kk = 0; kk < FN; kk += 32) {
      v16bf a = load_A_lds(Ha, HS_STRIDE, lane, kk);
      __builtin_prefetch(W2e + (size_t)(ffc + kk + 32) * D_MODEL + on0, 0, 1);
#pragma unroll
      for (int g = 0; g < 2; ++g) {      // two groups of 8 to bound live temps
#pragma unroll
        for (int t = 0; t < 8; ++t) {
          const int tt = g * 8 + t;
          v16bf b = load_B_g(
              W2e + (size_t)(ffc + kk) * D_MODEL + (on0 + tt * 16), D_MODEL,
              lane);
          oacc[tt] = __builtin_amdgcn_wmma_f32_16x16x32_bf16(
              false, a, false, b, (short)0, oacc[tt], false, false);
        }
      }
    }
  }

  // ---- Write gated output (per-row, slot-separated => deterministic) ------
  const int n  = lane & 15;
  const int mb = mrow0 + ((lane >> 4) << 3);
#pragma unroll
  for (int t = 0; t < 16; ++t) {
    const int col = on0 + t * 16 + n;
#pragma unroll
    for (int r = 0; r < 8; ++r) {
      const int m = mb + r;
      if (row0 + m < cnt) {
        const int packed = toks[m];       // token*2 + slot
        const int tok    = packed >> 1;
        const float g    = gates[packed]; // gates indexed exactly by packed
        const float val  = g * oacc[t][r];
        float* dst = (packed & 1) ? dst1 : dst0;
        const size_t oi = (size_t)tok * D_MODEL + col;
        if (atomic_mode) atomicAdd(dst + oi, val);
        else             dst[oi] = val;
      }
    }
  }
}

// ---- Kernel 3: combine slot partials --------------------------------------
__global__ __launch_bounds__(256) void moe_combine(
    const float4* __restrict__ p0, const float4* __restrict__ p1,
    float4* __restrict__ out) {
  int i = blockIdx.x * 256 + threadIdx.x;
  float4 a = p0[i], b = p1[i];
  out[i] = make_float4(a.x + b.x, a.y + b.y, a.z + b.z, a.w + b.w);
}

// ---- Host launcher ---------------------------------------------------------
extern "C" void kernel_launch(void* const* d_in, const int* in_sizes, int n_in,
                              void* d_out, int out_size, void* d_ws,
                              size_t ws_size, hipStream_t stream) {
  const float* x      = (const float*)d_in[0]; // [4,2048,1024]
  const float* router = (const float*)d_in[1]; // [1024,8]
  const float* W1     = (const float*)d_in[2]; // [8,1024,4096]
  const float* W2     = (const float*)d_in[3]; // [8,4096,1024]
  float* out          = (float*)d_out;         // [4,2048,1024]

  char* ws = (char*)d_ws;
  const size_t gates_off = 256;
  const size_t tlist_off = gates_off + (size_t)T_TOK * 2 * sizeof(float);
  const size_t p_off =
      (tlist_off + (size_t)N_EXP * T_TOK * sizeof(int) + 255) & ~(size_t)255;
  const size_t p_bytes = (size_t)T_TOK * D_MODEL * sizeof(float);
  if (ws_size < p_off) return; // cannot even hold routing metadata

  int*   counts = (int*)(ws + 0);
  float* gates  = (float*)(ws + gates_off);
  int*   tlist  = (int*)(ws + tlist_off);

  moe_zero_counts<<<1, 32, 0, stream>>>(counts);
  moe_router_scatter<<<T_TOK / 8, 256, 0, stream>>>(x, router, tlist, counts,
                                                    gates);

  dim3 grid(T_TOK / TM, N_EXP); // over-provisioned; blocks past count exit
  if (ws_size >= p_off + 2 * p_bytes) {
    // Deterministic path: per-slot partial buffers, then combine.
    float* P0 = (float*)(ws + p_off);
    float* P1 = (float*)(ws + p_off + p_bytes);
    moe_expert_ffn<<<grid, 256, SMEM_BYTES, stream>>>(
        x, W1, W2, tlist, counts, gates, P0, P1, 0);
    moe_combine<<<(T_TOK * D_MODEL / 4) / 256, 256, 0, stream>>>(
        (const float4*)P0, (const float4*)P1, (float4*)out);
  } else {
    // Fallback: accumulate directly with float atomics.
    moe_zero_out<<<(T_TOK * D_MODEL) / 256, 256, 0, stream>>>(
        out, T_TOK * D_MODEL);
    moe_expert_ffn<<<grid, 256, SMEM_BYTES, stream>>>(
        x, W1, W2, tlist, counts, gates, out, out, 1);
  }
}
